// GeometricBilinear_71665824301380
// MI455X (gfx1250) — matrix operand order repaired and checked by reading, not compiled, not run on hardware
//
#include <hip/hip_runtime.h>

// ---------------------------------------------------------------------------
// GeometricBilinear for MI455X (gfx1250, wave32, WMMA + TDM).
//
// Per 32-token tile (one workgroup, 8 wave32s):
//   TDM : tensor_load_to_lds DMAs the 64KB f32 mv tile into LDS (TENSORcnt)
//   S1  : mv(32x512) @ WinT(512->1024) -> left|right|jl|jr (bf16 WMMA, f32 acc)
//         each B fragment feeds 2 M-tiles -> halves L2 weight traffic
//   S3  : per (token,channel) bilinear via K-streamed outer products:
//         A_j[m,k] = left[m,j]*right[m,k], B_j = T[:,j,:]  (32 accum steps)
//   S4  : hidden(32x512) @ WoutT(512->512) -> out_mv
//   S5  : out_s VALU epilogue.
// Fused weights precomputed into d_ws (bf16, WMMA-B tile layout) each call.
// ---------------------------------------------------------------------------

typedef __attribute__((ext_vector_type(16))) __bf16 v16bf;
typedef __attribute__((ext_vector_type(8))) float v8f;
typedef __attribute__((ext_vector_type(16))) unsigned short v16us;
typedef __attribute__((ext_vector_type(4))) unsigned int v4u;
typedef __attribute__((ext_vector_type(8))) int v8i;
typedef __attribute__((ext_vector_type(4))) int v4i;

union Frag {
  v16us v;
  uint4 q[2];
  unsigned short h[16];
};

__device__ __forceinline__ unsigned short f2bf(float f) {
  unsigned int u = __builtin_bit_cast(unsigned int, f);
  u += 0x7FFFu + ((u >> 16) & 1u);  // round-to-nearest-even
  return (unsigned short)(u >> 16);
}
__device__ __forceinline__ float bf2f(unsigned short h) {
  unsigned int u = ((unsigned int)h) << 16;
  return __builtin_bit_cast(float, u);
}
__device__ __forceinline__ v8f zero8() {
  v8f z;
#pragma unroll
  for (int i = 0; i < 8; ++i) z[i] = 0.0f;
  return z;
}
__device__ __forceinline__ v8f wmma_bf16(const Frag& a, const Frag& b, v8f c) {
  return __builtin_amdgcn_wmma_f32_16x16x32_bf16(
      false, __builtin_bit_cast(v16bf, a.v),
      false, __builtin_bit_cast(v16bf, b.v),
      (short)0, c, false, false);
}

// 1-D TDM descriptor: copy n_elems f32 from gaddr to LDS at lds_addr.
// D# group0: count=1 | lds_addr | global_addr[56:0] | type=2 ("image").
// D# group1: data_size=4B; tensor_dim0 = tile_dim0 = n_elems; dim1=1.
__device__ __forceinline__ void tdm_load_1d(unsigned lds_addr,
                                            unsigned long long gaddr,
                                            unsigned n_elems) {
  v4u g0;
  g0[0] = 1u;  // count=1, user mode, no gather
  g0[1] = lds_addr;
  g0[2] = (unsigned)(gaddr & 0xffffffffu);
  g0[3] = (unsigned)((gaddr >> 32) & 0x01ffffffu) | (2u << 30);
  v8i g1;
  g1[0] = (int)(2u << 16);                               // data_size = 4 bytes
  g1[1] = (int)((n_elems & 0xffffu) << 16);              // tensor_dim0[15:0]
  g1[2] = (int)(((n_elems >> 16) & 0xffffu) | (1u << 16));  // dim0 hi | dim1=1
  g1[3] = (int)((n_elems & 0xffffu) << 16);              // tile_dim0
  g1[4] = 0;                                             // tile_dim1/2 unused
  g1[5] = (int)n_elems;                                  // dim0_stride lo
  g1[6] = 0;
  g1[7] = 0;
  v4i z4 = {0, 0, 0, 0};
#if defined(__clang_major__) && (__clang_major__ >= 23)
  v8i z8 = {0, 0, 0, 0, 0, 0, 0, 0};
  __builtin_amdgcn_tensor_load_to_lds(g0, g1, z4, z4, z8, 0);
#else
  __builtin_amdgcn_tensor_load_to_lds(g0, g1, z4, z4, 0);
#endif
}

// ---- d_ws layout (bf16 element offsets) ----
#define WS_WIN 0          // 64 ntiles x 16 ktiles x 512  (1024x512 fused input W)
#define WS_WOUT 524288    // 32 ntiles x 16 ktiles x 512  (512x512 fused output W)
#define WS_TGP 786432     // 32 j x 2 ntiles x 512        (gp_table slices)
#define WS_TOUT 819200    // 32 j x 2 ntiles x 512        (outer_table slices)
#define WS_TOTAL 851968

// ---------------------------------------------------------------------------
// Prep: fused bf16 weight tiles in WMMA-B layout.
// B tile element (lane, h): n = lane&15, k = (h&7) + (h>=8 ? 16 : 0) + (lane>=16 ? 8 : 0).
// ---------------------------------------------------------------------------
__global__ __launch_bounds__(256) void prep_weights(
    const float* __restrict__ basis, const float* __restrict__ gp,
    const float* __restrict__ outer, const float* __restrict__ wl,
    const float* __restrict__ wr, const float* __restrict__ wjl,
    const float* __restrict__ wjr, const float* __restrict__ wout,
    unsigned short* __restrict__ ws) {
  int idx = blockIdx.x * 256 + threadIdx.x;
  if (idx >= WS_TOTAL) return;

  if (idx < WS_WOUT) {  // WinT
    int lh = idx & 511, tile = idx >> 9;
    int kt = tile & 15, ntile = tile >> 4;
    int lanev = lh >> 4, h = lh & 15;
    int n = lanev & 15, k = (h & 7) + ((h >= 8) ? 16 : 0) + ((lanev >= 16) ? 8 : 0);
    int ng = ntile * 16 + n, kg = kt * 32 + k;
    int y = ng >> 5, i = ng & 31, x = kg >> 5, j = kg & 31;
    int br = y >> 3, c = y & 7;
    const float* w = (br == 0) ? wl : (br == 1) ? wr : (br == 2) ? wjl : wjr;
    const float* wp = w + (c * 16 + x) * 9;
    float acc = 0.f;
#pragma unroll
    for (int a = 0; a < 9; ++a) acc += wp[a] * basis[(a * 32 + i) * 32 + j];
    ws[idx] = f2bf(acc);
    return;
  }
  int i2 = idx - WS_WOUT;
  if (i2 < 262144) {  // WoutT
    int lh = i2 & 511, tile = i2 >> 9;
    int kt = tile & 15, ntile = tile >> 4;
    int lanev = lh >> 4, h = lh & 15;
    int n = lanev & 15, k = (h & 7) + ((h >= 8) ? 16 : 0) + ((lanev >= 16) ? 8 : 0);
    int ng = ntile * 16 + n, kg = kt * 32 + k;
    int y = ng >> 5, i = ng & 31, x = kg >> 5, j = kg & 31;
    const float* wp = wout + (y * 16 + x) * 9;
    float acc = 0.f;
#pragma unroll
    for (int a = 0; a < 9; ++a) acc += wp[a] * basis[(a * 32 + i) * 32 + j];
    ws[idx] = f2bf(acc);
    return;
  }
  int i3 = i2 - 262144;  // TgpT (32768) then ToutT (32768)
  const float* T = (i3 < 32768) ? gp : outer;
  int t = i3 & 32767;
  int lh = t & 511, tile = t >> 9;  // tile = j*2 + nt
  int nt = tile & 1, j = tile >> 1;
  int lanev = lh >> 4, h = lh & 15;
  int n = lanev & 15, k = (h & 7) + ((h >= 8) ? 16 : 0) + ((lanev >= 16) ? 8 : 0);
  int i = nt * 16 + n;
  ws[idx] = f2bf(T[(i * 32 + j) * 32 + k]);
}

// ---------------------------------------------------------------------------
// Main fused kernel. 256 threads (8 wave32s), 32 tokens per block.
// A fragment (16x32 bf16): M = lane&15; element e -> k = e + (e>=8 ? 8 : 0)
//   + (lane>=16 ? 8 : 0)   => two contiguous 8-half runs.
// C/D layout: VGPR r -> row r + (lane>=16 ? 8 : 0), col = lane&15.
// ---------------------------------------------------------------------------
__global__ __launch_bounds__(256) void gb_main(
    const float* __restrict__ mv, const float* __restrict__ refmv,
    const float* __restrict__ scal, const float* __restrict__ sLw,
    const float* __restrict__ sRw, const float* __restrict__ sJLw,
    const float* __restrict__ sJRw, const float* __restrict__ s2mv,
    const float* __restrict__ mv2s, const float* __restrict__ s2s,
    const float* __restrict__ bias, const unsigned short* __restrict__ winT,
    const unsigned short* __restrict__ woutT,
    const unsigned short* __restrict__ tgpT,
    const unsigned short* __restrict__ toutT, float* __restrict__ out,
    int Ntok) {
  // sLR: f32 TDM staging (64KB) during load, then left|right|jl|jr bf16.
  __shared__ __align__(16) unsigned short sLR[32 * 1024];  // 64 KB
  // sBuf: mv tile (bf16) during S1, hidden tile (bf16) during S3/S4.
  __shared__ __align__(16) unsigned short sBuf[32 * 512];  // 32 KB
  __shared__ float sScal[32 * 32];
  __shared__ float sH0[32 * 16];
  __shared__ float sRef[32];

  const int tid = threadIdx.x;
  const int lane = tid & 31;
  const int wv = tid >> 5;
  const size_t blk = blockIdx.x;

  // L2 prefetch of the fused weights (global_prefetch_b8 path).
  __builtin_prefetch(winT + (size_t)tid * 128, 0, 1);

  // ---- Load phase: TDM DMA of 32x512 f32 mv tile into LDS staging ----
  float* sStage = (float*)sLR;  // 16384 f32 == 64KB, aliases sLR
  if (wv == 0) {
    tdm_load_1d((unsigned)(size_t)(void*)sStage,
                (unsigned long long)(size_t)(const void*)(mv + blk * 16384),
                16384u);
  }
  // scalars + reference pseudoscalar coefficient (plain vector loads)
  ((float4*)sScal)[tid] = ((const float4*)scal)[blk * 256 + tid];
  if (tid < 32) sRef[tid] = refmv[(blk * 32 + tid) * 32 + 31];
  if (wv == 0) __builtin_amdgcn_s_wait_tensorcnt(0);
  __syncthreads();

  // ---- Convert staged f32 -> bf16 A-tile ----
#pragma unroll
  for (int it = 0; it < 16; ++it) {
    const int i4 = tid + it * 256;  // 0..4095 float4s
    float4 v = ((const float4*)sStage)[i4];
    uint2 u;
    u.x = (unsigned int)f2bf(v.x) | ((unsigned int)f2bf(v.y) << 16);
    u.y = (unsigned int)f2bf(v.z) | ((unsigned int)f2bf(v.w) << 16);
    *(uint2*)(&sBuf[i4 * 4]) = u;
  }
  __syncthreads();

  // ---- Stage 1: (32x512) @ WinT -> sLR (32x1024 bf16) ----
  {
    v8f acc[2][8];
#pragma unroll
    for (int s = 0; s < 2; ++s)
#pragma unroll
      for (int i = 0; i < 8; ++i) acc[s][i] = zero8();
    const unsigned short* arow0 = &sBuf[(lane & 15) * 512];
    const unsigned short* arow1 = &sBuf[(16 + (lane & 15)) * 512];
    const int abase = (lane >= 16) ? 8 : 0;
    for (int kt = 0; kt < 16; ++kt) {
      Frag a0, a1;
      a0.q[0] = *(const uint4*)(arow0 + kt * 32 + abase);
      a0.q[1] = *(const uint4*)(arow0 + kt * 32 + abase + 16);
      a1.q[0] = *(const uint4*)(arow1 + kt * 32 + abase);
      a1.q[1] = *(const uint4*)(arow1 + kt * 32 + abase + 16);
#pragma unroll
      for (int nti = 0; nti < 8; ++nti) {
        const int tile = (wv * 8 + nti) * 16 + kt;
        Frag b;
        const uint4* bp = (const uint4*)(winT + tile * 512 + lane * 16);
        b.q[0] = bp[0];
        b.q[1] = bp[1];
        acc[0][nti] = wmma_bf16(a0, b, acc[0][nti]);  // B reused: 2 M-tiles
        acc[1][nti] = wmma_bf16(a1, b, acc[1][nti]);
      }
    }
    const int n = lane & 15;
    const int rbase = (lane >= 16) ? 8 : 0;
    const float* sw4[4] = {sLw, sRw, sJLw, sJRw};
#pragma unroll
    for (int nti = 0; nti < 8; ++nti) {
      const int ng = (wv * 8 + nti) * 16 + n;
      const bool blade0 = ((ng & 31) == 0);
      const int q = ng >> 5;
      const float* sw = sw4[q >> 3] + (q & 7) * 32;
#pragma unroll
      for (int s = 0; s < 2; ++s)
#pragma unroll
        for (int r = 0; r < 8; ++r) {
          const int m = s * 16 + r + rbase;
          float vv = acc[s][nti][r];
          if (blade0) {
            float sacc = 0.f;
            for (int t = 0; t < 32; ++t) sacc += sScal[m * 32 + t] * sw[t];
            vv += sacc;
          }
          sLR[m * 1024 + ng] = f2bf(vv);
        }
    }
  }
  __syncthreads();

  // ---- Stage 3: bilinear gp/join via K-streamed outer products ----
  for (int s = 0; s < 2; ++s) {
    const int mt = wv * 2 + s;  // M-tile of (token*8 + channel) rows
    const int r = mt * 16 + (lane & 15);
    const int tok = r >> 3;
    const int c = r & 7;
    const unsigned short* rowp = &sLR[tok * 1024];
    const int base = (lane >= 16) ? 8 : 0;
    Frag rG, rJ;
    rG.q[0] = *(const uint4*)(rowp + 256 + c * 32 + base);
    rG.q[1] = *(const uint4*)(rowp + 256 + c * 32 + base + 16);
    rJ.q[0] = *(const uint4*)(rowp + 768 + c * 32 + base);
    rJ.q[1] = *(const uint4*)(rowp + 768 + c * 32 + base + 16);
    float fG[16], fJ[16];
#pragma unroll
    for (int e = 0; e < 16; ++e) {
      fG[e] = bf2f(rG.h[e]);
      fJ[e] = bf2f(rJ.h[e]);
    }
    v8f aGP[2], aJN[2];
#pragma unroll
    for (int i = 0; i < 2; ++i) {
      aGP[i] = zero8();
      aJN[i] = zero8();
    }
    for (int j = 0; j < 32; ++j) {
      const float lvG = bf2f(rowp[c * 32 + j]);
      const float lvJ = bf2f(rowp[512 + c * 32 + j]);
      Frag aG, aJ;
#pragma unroll
      for (int e = 0; e < 16; ++e) {
        aG.h[e] = f2bf(lvG * fG[e]);
        aJ.h[e] = f2bf(lvJ * fJ[e]);
      }
#pragma unroll
      for (int nt = 0; nt < 2; ++nt) {
        Frag bG, bJ;
        const uint4* pg = (const uint4*)(tgpT + (j * 2 + nt) * 512 + lane * 16);
        bG.q[0] = pg[0];
        bG.q[1] = pg[1];
        const uint4* po =
            (const uint4*)(toutT + (j * 2 + nt) * 512 + lane * 16);
        bJ.q[0] = po[0];
        bJ.q[1] = po[1];
        aGP[nt] = wmma_bf16(aG, bG, aGP[nt]);
        aJN[nt] = wmma_bf16(aJ, bJ, aJN[nt]);
      }
    }
    const int n = lane & 15;
    const int rbase = (lane >= 16) ? 8 : 0;
#pragma unroll
    for (int nt = 0; nt < 2; ++nt) {
      const int ib = nt * 16 + n;  // blade index
#pragma unroll
      for (int rr = 0; rr < 8; ++rr) {
        const int rg = mt * 16 + rr + rbase;
        const int tk = rg >> 3;
        const int ch = rg & 7;
        const float g = aGP[nt][rr];
        const float jn = aJN[nt][rr] * sRef[tk];
        sBuf[tk * 512 + ch * 32 + ib] = f2bf(g);
        sBuf[tk * 512 + (8 + ch) * 32 + ib] = f2bf(jn);
        if (ib == 0) {
          sH0[tk * 16 + ch] = g;
          sH0[tk * 16 + 8 + ch] = jn;
        }
      }
    }
  }
  __syncthreads();

  // ---- Stage 4: hidden (32x512) @ WoutT -> out_mv ----
  {
    v8f acc[2][4];
#pragma unroll
    for (int s = 0; s < 2; ++s)
#pragma unroll
      for (int i = 0; i < 4; ++i) acc[s][i] = zero8();
    const unsigned short* arow0 = &sBuf[(lane & 15) * 512];
    const unsigned short* arow1 = &sBuf[(16 + (lane & 15)) * 512];
    const int abase = (lane >= 16) ? 8 : 0;
    for (int kt = 0; kt < 16; ++kt) {
      Frag a0, a1;
      a0.q[0] = *(const uint4*)(arow0 + kt * 32 + abase);
      a0.q[1] = *(const uint4*)(arow0 + kt * 32 + abase + 16);
      a1.q[0] = *(const uint4*)(arow1 + kt * 32 + abase);
      a1.q[1] = *(const uint4*)(arow1 + kt * 32 + abase + 16);
#pragma unroll
      for (int nti = 0; nti < 4; ++nti) {
        const int tile = (wv * 4 + nti) * 16 + kt;
        Frag b;
        const uint4* bp = (const uint4*)(woutT + tile * 512 + lane * 16);
        b.q[0] = bp[0];
        b.q[1] = bp[1];
        acc[0][nti] = wmma_bf16(a0, b, acc[0][nti]);
        acc[1][nti] = wmma_bf16(a1, b, acc[1][nti]);
      }
    }
    const int n = lane & 15;
    const int rbase = (lane >= 16) ? 8 : 0;
#pragma unroll
    for (int nti = 0; nti < 4; ++nti) {
      const int ng = (wv * 4 + nti) * 16 + n;
      const bool blade0 = ((ng & 31) == 0);
      const int y = ng >> 5;
#pragma unroll
      for (int s = 0; s < 2; ++s)
#pragma unroll
        for (int r0 = 0; r0 < 8; ++r0) {
          const int m = s * 16 + r0 + rbase;
          float vv = acc[s][nti][r0];
          if (blade0) {
            float sacc = 0.f;
            for (int t = 0; t < 32; ++t)
              sacc += sScal[m * 32 + t] * s2mv[y * 32 + t];
            vv += sacc;
          }
          out[(blk * 32 + m) * 512 + ng] = vv;
        }
    }
  }

  // ---- Stage 5: out_s (sH0 from S3; barrier above covers it) ----
  {
    const int m = tid >> 3;  // 0..31
    float* outS = out + (size_t)Ntok * 512;
#pragma unroll
    for (int oo = tid & 7; oo < 32; oo += 8) {
      float v = bias[oo];
#pragma unroll
      for (int h = 0; h < 16; ++h) v += sH0[m * 16 + h] * mv2s[oo * 16 + h];
#pragma unroll
      for (int t = 0; t < 32; ++t) v += sScal[m * 32 + t] * s2s[oo * 32 + t];
      outS[(blk * 32 + m) * 32 + oo] = v;
    }
  }
}

extern "C" void kernel_launch(void* const* d_in, const int* in_sizes, int n_in,
                              void* d_out, int out_size, void* d_ws,
                              size_t ws_size, hipStream_t stream) {
  const float* mv = (const float*)d_in[0];
  const float* refmv = (const float*)d_in[1];
  const float* scal = (const float*)d_in[2];
  const float* basis = (const float*)d_in[3];
  const float* gp = (const float*)d_in[4];
  const float* outer = (const float*)d_in[5];
  const float* w_left = (const float*)d_in[6];
  const float* s_left = (const float*)d_in[7];
  const float* w_right = (const float*)d_in[8];
  const float* s_right = (const float*)d_in[9];
  const float* w_jleft = (const float*)d_in[10];
  const float* s_jleft = (const float*)d_in[11];
  const float* w_jright = (const float*)d_in[12];
  const float* s_jright = (const float*)d_in[13];
  const float* w_out = (const float*)d_in[14];
  const float* s2mv_out = (const float*)d_in[15];
  const float* mv2s_out = (const float*)d_in[16];
  const float* s2s_out = (const float*)d_in[17];
  const float* bias_s = (const float*)d_in[18];

  unsigned short* ws = (unsigned short*)d_ws;
  const int N = in_sizes[0] / 512;  // tokens

  prep_weights<<<(WS_TOTAL + 255) / 256, 256, 0, stream>>>(
      basis, gp, outer, w_left, w_right, w_jleft, w_jright, w_out, ws);

  gb_main<<<N / 32, 256, 0, stream>>>(
      mv, refmv, scal, s_left, s_right, s_jleft, s_jright, s2mv_out, mv2s_out,
      s2s_out, bias_s, ws + WS_WIN, ws + WS_WOUT, ws + WS_TGP, ws + WS_TOUT,
      (float*)d_out, N);
}
